// LocalRNN_79499844649264
// MI455X (gfx1250) — matrix operand order repaired and checked by reading, not compiled, not run on hardware
//
#include <hip/hip_runtime.h>
#include <hip/hip_bf16.h>

// ---------------------------------------------------------------------------
// Local GRU (chunked, h0 = 0 per chunk) for MI455X / gfx1250.
// B=32, S=2048, I=H=256, KSIZE=16  ->  4096 independent length-16 recurrences.
//
//  * bf16 WMMA (v_wmma_f32_16x16x32_bf16), fp32 accumulate, fp32 gates/state.
//  * 1 WG = 32 chunk-rows (two 16-row M-tiles), 16 waves; wave w owns output
//    columns 16w..16w+15 and gate column-tiles {w, w+16, w+32} (wave-local
//    gating on the WMMA C layout).
//  * Weights pre-swizzled once into d_ws (bf16, B-operand lane layout,
//    32B contiguous per lane per tile), restreamed from L2 each step.
//    An asm barrier on a 64-bit *integer* base blocks LICM/CSE across t
//    (no spills) while explicit addrspace(1) derefs keep global_load_b128.
//  * kt=0 peeled with literal-zero C so the first WMMAs use inline SRC2=0.
//  * x_t and h staged in LDS in the A-operand lane layout: one v16bf LDS
//    load per A tile (2x ds_load_b128, zero register shuffling).
// ---------------------------------------------------------------------------

typedef __bf16 bf16_t;
typedef __attribute__((ext_vector_type(16))) __bf16 v16bf;
typedef __attribute__((ext_vector_type(8)))  float  v8f;
typedef __attribute__((address_space(1))) const v16bf* g16p;

#define HDIM   256
#define KS     16
#define NROWS  4096                 // B*S/KS independent recurrences
#define MROWS  32                   // chunk-rows per workgroup (2 M-tiles)
#define NT     48                   // 768/16 column tiles
#define KT     8                    // 256/32 k tiles
#define WBLK   512                  // bf16 elems per (ntile,kt) block: 32 lanes * 16
#define WMAT   (NT * KT * WBLK)     // 196608 bf16 elems per weight matrix
#define ABUF   (2 * KT * WBLK)      // A-swizzled LDS buffer: 2 M-tiles x 8 kt x 512
#define GATE_B (2ull * 16 * KT * WBLK)  // byte stride between gate column groups

// --- weight swizzle: fp32 [768,256] row-major -> bf16 B-operand tile layout.
// B (32x16 bf16) lane layout: lane L holds N = L%16, K = (L/16)*8 + (e&7) + (e/8)*16,
// stored contiguously: dst[((ntile*8+kt)*32 + lane)*16 + e].
__global__ __launch_bounds__(256) void swz_kernel(const float* __restrict__ Wih,
                                                  const float* __restrict__ Whh,
                                                  bf16_t* __restrict__ dst) {
  int idx = blockIdx.x * 256 + threadIdx.x;        // 0 .. 2*WMAT-1
  const float* src = (idx < WMAT) ? Wih : Whh;
  int r     = (idx < WMAT) ? idx : idx - WMAT;
  int blk   = r >> 9;                              // (ntile*8 + kt)
  int le    = r & 511;
  int lane  = le >> 4;
  int e     = le & 15;
  int ntile = blk >> 3;
  int kt    = blk & 7;
  int row   = ntile * 16 + (lane & 15);            // weight row (output col)
  int k     = kt * 32 + (lane >> 4) * 8 + (e & 7) + ((e >> 3) << 4);      // weight col (k index)
  dst[idx] = (bf16_t)src[row * HDIM + k];
}

__device__ __forceinline__ float sigmoid_f(float x) {
  return 1.0f / (1.0f + __expf(-x));
}
__device__ __forceinline__ float tanh_f(float x) {
  return 1.0f - 2.0f / (1.0f + __expf(2.0f * x));
}

// A-operand LDS index for logical (row m in 0..31, k in 0..255).
__device__ __forceinline__ int a_idx(int m, int k) {
  int mtile = m >> 4;
  int lane  = (m & 15) + (((k >> 3) & 1) << 4);
  int e     = (k & 7) + (((k >> 4) & 1) << 3);
  return (((mtile << 3) + (k >> 5)) << 9) + (lane << 4) + e;
}

#define WMMA_BF16(A, B, C) \
  __builtin_amdgcn_wmma_f32_16x16x32_bf16(false, (A), false, (B), (short)0, (C), false, false)

__global__ __launch_bounds__(512) void gru_fused_kernel(
    const float* __restrict__ x,      // flat [4096,16,256] fp32
    const float* __restrict__ b_ih,   // [768]
    const float* __restrict__ b_hh,   // [768]
    const bf16_t* Wsw,                // swizzled bf16 weights: [WMAT ih | WMAT hh]
    float* __restrict__ out) {        // flat [4096,16,256] fp32
  __shared__ bf16_t xs [ABUF];
  __shared__ bf16_t hsm[ABUF];

  const int tid     = threadIdx.x;
  const int lane    = tid & 31;
  const int wv      = tid >> 5;                 // wave id 0..15 == column tile
  const int col     = (wv << 4) + (lane & 15);  // this lane's H column 0..255
  const int rowbase = blockIdx.x * MROWS;       // chunk rows of this WG

  for (int i = tid; i < ABUF; i += 512) hsm[i] = (bf16_t)0.0f;

  const float bir  = b_ih[col];
  const float biz  = b_ih[HDIM + col];
  const float bin_ = b_ih[2 * HDIM + col];
  const float bhr  = b_hh[col];
  const float bhz  = b_hh[HDIM + col];
  const float bhn  = b_hh[2 * HDIM + col];

  unsigned long long wbase0 = (unsigned long long)Wsw;

  v8f h0 = {0.f,0.f,0.f,0.f,0.f,0.f,0.f,0.f};
  v8f h1 = {0.f,0.f,0.f,0.f,0.f,0.f,0.f,0.f};

  const int mbrow = (lane >> 4) << 3;           // C-layout row base (0 or 8)
  // h write-back constants (A-layout coords of k == col)
  const int h_lanehi = ((col >> 3) & 1) << 4;
  const int h_e      = (col & 7) + (((col >> 4) & 1) << 3);
  const int h_kt     = col >> 5;

  __syncthreads();

  for (int t = 0; t < KS; ++t) {
    // Opaque 64-bit weight base per step: blocks LICM/CSE across t (which
    // previously hoisted all B tiles and spilled) without changing memory
    // semantics; explicit addrspace(1) derefs keep global_load_b128.
    unsigned long long wb = wbase0;
    asm volatile("" : "+s"(wb));
    const unsigned long long ihb = wb;
    const unsigned long long hhb = wb + 2ull * WMAT;

    // ---- stage x_t: MROWS x 256 fp32 -> bf16 LDS in A-operand layout
#pragma unroll
    for (int rep = 0; rep < 4; ++rep) {
      int f = ((rep << 9) + tid) << 2;       // flat float index, 4 per thread
      int m = f >> 8;
      int k = f & 255;
      const float4 v = *reinterpret_cast<const float4*>(
          x + ((size_t)(rowbase + m) * KS + t) * HDIM + k);
      bf16_t* p = &xs[a_idx(m, k)];          // 4 consecutive elems, 8B aligned
      p[0] = (bf16_t)v.x; p[1] = (bf16_t)v.y;
      p[2] = (bf16_t)v.z; p[3] = (bf16_t)v.w;
    }
    __syncthreads();

    const v8f z8 = {0.f,0.f,0.f,0.f,0.f,0.f,0.f,0.f};
    v8f air0, air1, aiz0, aiz1, ain0, ain1;
    v8f ahr0, ahr1, ahz0, ahz1, ahn0, ahn1;

    // ---- kt = 0 peeled: literal-zero C -> inline SRC2=0 in the WMMAs
    {
      const int a0 = lane << 4;
      const int a1 = (8 << 9) + (lane << 4);
      v16bf Ax0 = *reinterpret_cast<const v16bf*>(&xs [a0]);
      v16bf Ax1 = *reinterpret_cast<const v16bf*>(&xs [a1]);
      v16bf Ah0 = *reinterpret_cast<const v16bf*>(&hsm[a0]);
      v16bf Ah1 = *reinterpret_cast<const v16bf*>(&hsm[a1]);

      const unsigned long long bb = (unsigned long long)((((wv << 3) + 0) << 9) + (lane << 4)) * 2ull;
      v16bf Bir = *(g16p)(ihb + bb);
      v16bf Biz = *(g16p)(ihb + bb + GATE_B);
      v16bf Bin = *(g16p)(ihb + bb + 2 * GATE_B);
      v16bf Bhr = *(g16p)(hhb + bb);
      v16bf Bhz = *(g16p)(hhb + bb + GATE_B);
      v16bf Bhn = *(g16p)(hhb + bb + 2 * GATE_B);

      air0 = WMMA_BF16(Ax0, Bir, z8);  air1 = WMMA_BF16(Ax1, Bir, z8);
      aiz0 = WMMA_BF16(Ax0, Biz, z8);  aiz1 = WMMA_BF16(Ax1, Biz, z8);
      ain0 = WMMA_BF16(Ax0, Bin, z8);  ain1 = WMMA_BF16(Ax1, Bin, z8);
      ahr0 = WMMA_BF16(Ah0, Bhr, z8);  ahr1 = WMMA_BF16(Ah1, Bhr, z8);
      ahz0 = WMMA_BF16(Ah0, Bhz, z8);  ahz1 = WMMA_BF16(Ah1, Bhz, z8);
      ahn0 = WMMA_BF16(Ah0, Bhn, z8);  ahn1 = WMMA_BF16(Ah1, Bhn, z8);
    }

#pragma clang loop unroll(disable)
    for (int kt = 1; kt < KT; ++kt) {
      // A tiles: one v16bf LDS load each (2x ds_load_b128, no shuffles)
      const int a0 = (kt << 9) + (lane << 4);
      const int a1 = ((8 + kt) << 9) + (lane << 4);
      v16bf Ax0 = *reinterpret_cast<const v16bf*>(&xs [a0]);
      v16bf Ax1 = *reinterpret_cast<const v16bf*>(&xs [a1]);
      v16bf Ah0 = *reinterpret_cast<const v16bf*>(&hsm[a0]);
      v16bf Ah1 = *reinterpret_cast<const v16bf*>(&hsm[a1]);

      // B tiles: contiguous 32B per lane (2x global_load_b128, L2-resident)
      const unsigned long long bb = (unsigned long long)((((wv << 3) + kt) << 9) + (lane << 4)) * 2ull;
      v16bf Bir = *(g16p)(ihb + bb);
      v16bf Biz = *(g16p)(ihb + bb + GATE_B);
      v16bf Bin = *(g16p)(ihb + bb + 2 * GATE_B);
      v16bf Bhr = *(g16p)(hhb + bb);
      v16bf Bhz = *(g16p)(hhb + bb + GATE_B);
      v16bf Bhn = *(g16p)(hhb + bb + 2 * GATE_B);

      // Prefetch next k-tile's weight blocks toward the WGP (no VGPR cost).
      if (kt + 1 < KT) {
        const unsigned long long bn = bb + 2ull * WBLK;
        __builtin_prefetch((const bf16_t*)(ihb + bn), 0, 3);
        __builtin_prefetch((const bf16_t*)(ihb + bn + GATE_B), 0, 3);
        __builtin_prefetch((const bf16_t*)(ihb + bn + 2 * GATE_B), 0, 3);
        __builtin_prefetch((const bf16_t*)(hhb + bn), 0, 3);
        __builtin_prefetch((const bf16_t*)(hhb + bn + GATE_B), 0, 3);
        __builtin_prefetch((const bf16_t*)(hhb + bn + 2 * GATE_B), 0, 3);
      }

      air0 = WMMA_BF16(Ax0, Bir, air0);  air1 = WMMA_BF16(Ax1, Bir, air1);
      aiz0 = WMMA_BF16(Ax0, Biz, aiz0);  aiz1 = WMMA_BF16(Ax1, Biz, aiz1);
      ain0 = WMMA_BF16(Ax0, Bin, ain0);  ain1 = WMMA_BF16(Ax1, Bin, ain1);
      ahr0 = WMMA_BF16(Ah0, Bhr, ahr0);  ahr1 = WMMA_BF16(Ah1, Bhr, ahr1);
      ahz0 = WMMA_BF16(Ah0, Bhz, ahz0);  ahz1 = WMMA_BF16(Ah1, Bhz, ahz1);
      ahn0 = WMMA_BF16(Ah0, Bhn, ahn0);  ahn1 = WMMA_BF16(Ah1, Bhn, ahn1);
    }

    // ---- fp32 gate math on the C layout; row m = mbrow+e (+16 for tile 1)
#pragma unroll
    for (int e = 0; e < 8; ++e) {
      float r  = sigmoid_f(air0[e] + bir + ahr0[e] + bhr);
      float z  = sigmoid_f(aiz0[e] + biz + ahz0[e] + bhz);
      float n  = tanh_f(ain0[e] + bin_ + r * (ahn0[e] + bhn));
      float hn = (1.0f - z) * n + z * h0[e];
      h0[e] = hn;
      const int m = mbrow + e;
      out[((size_t)(rowbase + m) * KS + t) * HDIM + col] = hn;
      hsm[(h_kt << 9) + (((m & 15) + h_lanehi) << 4) + h_e] = (bf16_t)hn;
    }
#pragma unroll
    for (int e = 0; e < 8; ++e) {
      float r  = sigmoid_f(air1[e] + bir + ahr1[e] + bhr);
      float z  = sigmoid_f(aiz1[e] + biz + ahz1[e] + bhz);
      float n  = tanh_f(ain1[e] + bin_ + r * (ahn1[e] + bhn));
      float hn = (1.0f - z) * n + z * h1[e];
      h1[e] = hn;
      const int m = mbrow + e;
      out[((size_t)(rowbase + 16 + m) * KS + t) * HDIM + col] = hn;
      hsm[((8 + h_kt) << 9) + (((m & 15) + h_lanehi) << 4) + h_e] = (bf16_t)hn;
    }
    __syncthreads();   // h visible for next step; x stage may be overwritten
  }
}

extern "C" void kernel_launch(void* const* d_in, const int* in_sizes, int n_in,
                              void* d_out, int out_size, void* d_ws, size_t ws_size,
                              hipStream_t stream) {
  const float* x    = (const float*)d_in[0];   // [32,2048,256]
  const float* W_ih = (const float*)d_in[1];   // [768,256]
  const float* W_hh = (const float*)d_in[2];   // [768,256]
  const float* b_ih = (const float*)d_in[3];   // [768]
  const float* b_hh = (const float*)d_in[4];   // [768]
  // d_in[5] = ksize (== 16, hardcoded)
  (void)in_sizes; (void)n_in; (void)out_size; (void)ws_size;

  bf16_t* Wsw = (bf16_t*)d_ws;                 // needs 2*WMAT*2 = 768 KB scratch

  // 1) swizzle weights fp32 -> bf16 B-operand layout (stays resident in L2)
  swz_kernel<<<(2 * WMAT) / 256, 256, 0, stream>>>(W_ih, W_hh, Wsw);

  // 2) fused local-GRU: 128 WGs (32 chunk-rows each) x 512 threads (16 waves)
  gru_fused_kernel<<<NROWS / MROWS, 512, 0, stream>>>(x, b_ih, b_hh, Wsw,
                                                      (float*)d_out);
}
